// Transformer_19430432047612
// MI455X (gfx1250) — matrix operand order repaired and checked by reading, not compiled, not run on hardware
//
#include <hip/hip_runtime.h>

// ---------------------------------------------------------------------------
// Shapes (fixed by the reference)
// ---------------------------------------------------------------------------
#define LAY 2
#define DM  256
#define NH  8
#define DH  32
#define CF  2
#define FFD 2048
#define BB  10
#define TT  1600
#define SS  802           // 1 null token + 801 compressed tokens
#define BTD (BB*TT*DM)    // 4,096,000

typedef _Float16 f16;
typedef __attribute__((ext_vector_type(16))) _Float16 v16h;
typedef __attribute__((ext_vector_type(8)))  _Float16 h8;
typedef __attribute__((ext_vector_type(8)))  float    v8f;
typedef __attribute__((ext_vector_type(4)))  unsigned int u32x4;
typedef __attribute__((ext_vector_type(4)))  int          i32x4;
typedef __attribute__((ext_vector_type(8)))  int          i32x8;

struct HalfVec { h8 lo, hi; };   // 32B -> bit_cast to v16h

static __device__ __forceinline__ float ld_f16(const f16* p) { return (float)*p; }

// per-lane async global->LDS b128 (ASYNCcnt), no VGPR round-trip
static __device__ __forceinline__ void async_ld_b128(unsigned lds_byte_addr,
                                                     const void* gptr) {
  asm volatile("global_load_async_to_lds_b128 %0, %1, off"
               :: "v"(lds_byte_addr), "v"(gptr) : "memory");
}
static __device__ __forceinline__ void wait_asynccnt0() {
  asm volatile("s_wait_asynccnt 0x0" ::: "memory");
}

// ---------------------------------------------------------------------------
// Tiled f16 WMMA GEMM:  C[M,N] = A[M,K] @ W[N,K]^T (+bias) (+relu)
// A, W row-major f16.  Output either f32 (Cf) or f16 (Ch).
// Block: 256 thr = 8 waves; tile 128x128; wave = 32(M) x 64(N) = 2x4 WMMA tiles.
// Staging: A tile via per-lane async global->LDS loads (ASYNCcnt);
//          B tile via one TDM tensor_load_to_lds descriptor per K-step
//          (TENSORcnt), using D# LDS padding to produce the 112B row stride.
// Double-buffered: stage tile k+1 while computing WMMAs on tile k.
// M % 128 == 0, N % 128 == 0, K % 32 == 0 for all calls here.
// ---------------------------------------------------------------------------
__global__ __launch_bounds__(256) void k_gemm_f16(
    const f16* __restrict__ A, const f16* __restrict__ W,
    const float* __restrict__ bias, float* __restrict__ Cf, f16* __restrict__ Ch,
    int M, int N, int K, int relu)
{
  __shared__ _Float16 As[2][128][56];   // 112B rows: 16B-aligned, bank-spread
  __shared__ _Float16 Bs[2][128][56];

  const int t    = threadIdx.x;
  const int lane = t & 31;
  const int wave = t >> 5;
  const int wm   = wave & 3;     // M sub-tile 0..3  (32 rows each)
  const int wn   = wave >> 2;    // N sub-tile 0..1  (64 cols each)
  const int bM   = blockIdx.y * 128;
  const int bN   = blockIdx.x * 128;

  const int mrow = lane & 15;
  const int ksel = lane >> 4;    // 0/1 selects K-half (k = 16*ksel + 0..15), same order in A and B

  // --- A tile: 512 x 16B chunks, 2 per thread, async to LDS ---
  auto stage_A = [&](int buf, int kk) {
#pragma unroll
    for (int i = 0; i < 2; ++i) {
      int c = i * 256 + t;
      int r = c >> 2, q = c & 3;
      unsigned lds = (unsigned)(uintptr_t)&As[buf][r][q * 8];
      async_ld_b128(lds, &A[(size_t)(bM + r) * K + kk + q * 8]);
    }
  };

  // --- B tile: one TDM descriptor moves 128 rows x 64B, padded to 112B rows ---
  auto stage_B = [&](int buf, int kk) {
    if (wave != 0) return;
    unsigned lds = (unsigned)(uintptr_t)&Bs[buf][0][0];
    unsigned long long ga =
        (unsigned long long)(uintptr_t)(W + (size_t)bN * K + kk);
    unsigned td0 = (unsigned)(K >> 1);   // tensor dim0 (row length, dwords)
    unsigned td1 = (unsigned)N;          // tensor dim1 (rows)
    u32x4 g0 = { 1u,                                   // count=1, user mode
                 lds,                                  // lds_addr
                 (unsigned)ga,                         // global_addr[31:0]
                 (unsigned)((ga >> 32) & 0x01ffffffu)  // global_addr[56:32]
                   | (2u << 30) };                     // type=2 (image)
    unsigned w0 = (2u << 16)    // data_size = 4B elements
                | (1u << 20)    // pad_enable
                | (3u << 22)    // pad_interval: every 16 dwords (64B)
                | (11u << 25);  // pad_amount: 12 dwords (48B) -> 112B stride
    unsigned w1 = (td0 & 0xffffu) << 16;
    unsigned w2 = (td0 >> 16) | ((td1 & 0xffffu) << 16);
    unsigned w3 = (td1 >> 16) | (16u << 16);   // tile_dim0 = 16 dwords (64B)
    unsigned w4 = 128u;                        // tile_dim1 = 128 rows
    unsigned w5 = (unsigned)(K >> 1);          // tensor_dim0_stride (dwords)
    i32x8 g1 = { (int)w0, (int)w1, (int)w2, (int)w3,
                 (int)w4, (int)w5, 0, 0 };
    i32x4 g2 = { 0, 0, 0, 0 };
    i32x4 g3 = { 0, 0, 0, 0 };
#if defined(__clang_major__) && __clang_major__ >= 23
    i32x8 g4 = { 0, 0, 0, 0, 0, 0, 0, 0 };
    __builtin_amdgcn_tensor_load_to_lds(g0, g1, g2, g3, g4, 0);
#else
    __builtin_amdgcn_tensor_load_to_lds(g0, g1, g2, g3, 0);
#endif
  };

  auto wait_stage = [&]() {
    wait_asynccnt0();
    if (wave == 0) __builtin_amdgcn_s_wait_tensorcnt(0);
    __syncthreads();
  };

  v8f acc[2][4];
#pragma unroll
  for (int i = 0; i < 2; ++i)
#pragma unroll
    for (int j = 0; j < 4; ++j) acc[i][j] = (v8f)0.0f;

  const int nk = K >> 5;
  stage_B(0, 0);
  stage_A(0, 0);
  wait_stage();

  for (int kt = 0; kt < nk; ++kt) {
    const int cur = kt & 1;
    if (kt + 1 < nk) {               // stage next tile into the other buffer
      stage_B(cur ^ 1, (kt + 1) * 32);
      stage_A(cur ^ 1, (kt + 1) * 32);
    }

    // --- fragments & WMMAs on current buffer ---
    v16h bfrag[4];
#pragma unroll
    for (int nt = 0; nt < 4; ++nt) {
      int bn = wn * 64 + nt * 16 + mrow;
      HalfVec hv;
      hv.lo = *(const h8*)&Bs[cur][bn][ksel * 16];       // K = ksel*16 + 0..7
      hv.hi = *(const h8*)&Bs[cur][bn][ksel * 16 + 8];   // K = ksel*16 + 8..15
      bfrag[nt] = __builtin_bit_cast(v16h, hv);
    }
#pragma unroll
    for (int mt = 0; mt < 2; ++mt) {
      int am = wm * 32 + mt * 16 + mrow;
      HalfVec hv;
      hv.lo = *(const h8*)&As[cur][am][ksel * 16];       // K = ksel*16 + 0..7
      hv.hi = *(const h8*)&As[cur][am][ksel * 16 + 8];   // K = ksel*16 + 8..15
      v16h afrag = __builtin_bit_cast(v16h, hv);
#pragma unroll
      for (int nt = 0; nt < 4; ++nt) {
        acc[mt][nt] = __builtin_amdgcn_wmma_f32_16x16x32_f16(
            false, afrag, false, bfrag[nt], (short)0, acc[mt][nt], false, false);
      }
    }

    wait_stage();   // async/TDM waits overlap the WMMAs above
  }

  // --- epilogue: C layout  M = j + (lane>>4)*8,  N = lane&15 ---
#pragma unroll
  for (int mt = 0; mt < 2; ++mt)
#pragma unroll
    for (int nt = 0; nt < 4; ++nt)
#pragma unroll
      for (int j = 0; j < 8; ++j) {
        int row = bM + wm * 32 + mt * 16 + (lane >> 4) * 8 + j;
        int col = bN + wn * 64 + nt * 16 + (lane & 15);
        float v = acc[mt][nt][j];
        if (bias) v += bias[col];
        if (relu) v = fmaxf(v, 0.0f);
        size_t o = (size_t)row * N + col;
        if (Ch) Ch[o] = (f16)v; else Cf[o] = v;
      }
}

// ---------------------------------------------------------------------------
// Elementwise / small kernels
// ---------------------------------------------------------------------------

// [B,D,T] f32 -> [B,T,D] f32
__global__ void k_bdt_to_btd(const float* __restrict__ in, float* __restrict__ out) {
  int i = blockIdx.x * blockDim.x + threadIdx.x;
  if (i >= BTD) return;
  int d = i % DM; int bt = i / DM; int t = bt % TT; int b = bt / TT;
  out[i] = in[((size_t)b * DM + d) * TT + t];
}

// qkb = f16(x + pos); xb = f16(x)
__global__ void k_make_qk(const float* __restrict__ xf, const float* __restrict__ posf,
                          f16* __restrict__ qkb, f16* __restrict__ xb) {
  int i = blockIdx.x * blockDim.x + threadIdx.x;
  if (i >= BTD) return;
  float x = xf[i];
  qkb[i] = (f16)(x + posf[i]);
  xb[i]  = (f16)x;
}

__global__ void k_cvt_f16(const float* __restrict__ in, f16* __restrict__ out, int n) {
  int i = blockIdx.x * blockDim.x + threadIdx.x;
  if (i < n) out[i] = (f16)in[i];
}

// Grouped strided Conv1d over time (k=2, stride=2, groups=8), left-pad 2,
// prepend null token at s==0.  kv: [B,T,D] f16 -> out: [B,SS,D] f32
__global__ void k_compress(const f16* __restrict__ kv, const float* __restrict__ cw,
                           const float* __restrict__ cb, const float* __restrict__ nullkv,
                           float* __restrict__ out) {
  int i = blockIdx.x * blockDim.x + threadIdx.x;
  if (i >= BB * SS * DM) return;
  int d = i % DM; int bs = i / DM; int s = bs % SS; int b = bs / SS;
  if (s == 0) { out[i] = nullkv[b * DM + d]; return; }
  int tp = s - 1;                 // compressed time index 0..800
  int g  = d >> 5;                // head group
  float sum = cb[d];
#pragma unroll
  for (int j = 0; j < CF; ++j) {
    int t = 2 * tp + j - CF;      // source time (pad=2 zeros at the front)
    if (t >= 0) {
      const f16*  kr = kv + ((size_t)b * TT + t) * DM + g * DH;
      const float* w = cw + (size_t)d * DH * CF + j;
      for (int c = 0; c < DH; ++c) sum += w[c * CF] * ld_f16(&kr[c]);
    }
  }
  out[i] = sum;
}

// M[b,h] = scale * Kc[b,:,h]^T @ Vc[b,:,h]   (32x32 per (b,h), reduce over SS)
__global__ void k_ktv(const float* __restrict__ kc, const float* __restrict__ vc,
                      float* __restrict__ Mm) {
  int i = blockIdx.x * blockDim.x + threadIdx.x;
  if (i >= BB * NH * DH * DH) return;
  int d = i & 31; int e = (i >> 5) & 31; int h = (i >> 10) & 7; int b = i >> 13;
  const float* kcol = kc + (size_t)b * SS * DM + h * DH + e;
  const float* vcol = vc + (size_t)b * SS * DM + h * DH + d;
  float sum = 0.0f;
  for (int s = 0; s < SS; ++s) sum += kcol[(size_t)s * DM] * vcol[(size_t)s * DM];
  Mm[i] = sum * 0.0625f;          // D^-0.5 = 1/16
}

// attn[b,t,h,:] = q[b,t,h,:] @ M[b,h]   -> f16
__global__ void k_apply_m(const f16* __restrict__ q, const float* __restrict__ Mm,
                          f16* __restrict__ attnb) {
  int i = blockIdx.x * blockDim.x + threadIdx.x;
  if (i >= BTD) return;
  int c = i & 255; int bt = i >> 8; int b = bt / TT;
  int h = c >> 5, d = c & 31;
  const f16*  qr = q  + (size_t)bt * DM + h * DH;
  const float* m = Mm + (((size_t)b * NH + h) * DH) * DH + d;
  float sum = 0.0f;
#pragma unroll
  for (int e = 0; e < DH; ++e) sum += ld_f16(&qr[e]) * m[(size_t)e * DH];
  attnb[i] = (f16)sum;
}

// x = LN(x + resid) row-wise over D=256; one wave (32 lanes) per row.
__global__ __launch_bounds__(256) void k_add_ln(
    const float* __restrict__ resid, float* __restrict__ xf, f16* __restrict__ xh,
    const float* __restrict__ gam, const float* __restrict__ bet, int rows)
{
  int wave = threadIdx.x >> 5, lane = threadIdx.x & 31;
  int row = blockIdx.x * 8 + wave;
  if (row >= rows) return;
  const float* xr = xf    + (size_t)row * DM;
  const float* rr = resid + (size_t)row * DM;
  float v[8], s = 0.0f, s2 = 0.0f;
#pragma unroll
  for (int i = 0; i < 8; ++i) {
    int c = lane + 32 * i;
    v[i] = xr[c] + rr[c];
    s += v[i]; s2 += v[i] * v[i];
  }
#pragma unroll
  for (int off = 16; off > 0; off >>= 1) {
    s  += __shfl_xor(s,  off, 32);
    s2 += __shfl_xor(s2, off, 32);
  }
  float mean = s * (1.0f / DM);
  float var  = s2 * (1.0f / DM) - mean * mean;
  float inv  = rsqrtf(var + 1e-5f);
#pragma unroll
  for (int i = 0; i < 8; ++i) {
    int c = lane + 32 * i;
    float y = (v[i] - mean) * inv * gam[c] + bet[c];
    xf[(size_t)row * DM + c] = y;
    if (xh) xh[(size_t)row * DM + c] = (f16)y;
  }
}

// [B,T,D] f32 -> [B,D,T] f32 (final output)
__global__ void k_btd_to_bdt(const float* __restrict__ in, float* __restrict__ out) {
  int i = blockIdx.x * blockDim.x + threadIdx.x;
  if (i >= BTD) return;
  int t = i % TT; int bd = i / TT; int d = bd % DM; int b = bd / DM;
  out[i] = in[((size_t)b * TT + t) * DM + d];
}

// ---------------------------------------------------------------------------
// Host driver
// ---------------------------------------------------------------------------
extern "C" void kernel_launch(void* const* d_in, const int* in_sizes, int n_in,
                              void* d_out, int out_size, void* d_ws, size_t ws_size,
                              hipStream_t stream) {
  (void)in_sizes; (void)n_in; (void)out_size; (void)ws_size;
  const float* src    = (const float*)d_in[0];
  const float* pos    = (const float*)d_in[1];
  const float* Wq     = (const float*)d_in[2];
  const float* Wk     = (const float*)d_in[3];
  const float* Wv     = (const float*)d_in[4];
  const float* Wo     = (const float*)d_in[5];
  const float* bo     = (const float*)d_in[6];
  const float* conv_w = (const float*)d_in[7];
  const float* conv_b = (const float*)d_in[8];
  const float* null_k = (const float*)d_in[9];
  const float* null_v = (const float*)d_in[10];
  const float* ln1_g  = (const float*)d_in[11];
  const float* ln1_b  = (const float*)d_in[12];
  const float* ln2_g  = (const float*)d_in[13];
  const float* ln2_b  = (const float*)d_in[14];
  const float* W1     = (const float*)d_in[15];
  const float* b1     = (const float*)d_in[16];
  const float* W2     = (const float*)d_in[17];
  const float* b2     = (const float*)d_in[18];

  char* p = (char*)d_ws;
  auto alloc = [&](size_t bytes) -> void* {
    void* r = (void*)p;
    p += (bytes + 255) & ~(size_t)255;
    return r;
  };

  float* xf    = (float*)alloc((size_t)BTD * 4);
  float* posf  = (float*)alloc((size_t)BTD * 4);
  float* resid = (float*)alloc((size_t)BTD * 4);
  f16*   qkb   = (f16*)  alloc((size_t)BTD * 2);
  f16*   xb    = (f16*)  alloc((size_t)BTD * 2);
  f16*   qh    = (f16*)  alloc((size_t)BTD * 2);
  f16*   kh    = (f16*)  alloc((size_t)BTD * 2);
  f16*   vh    = (f16*)  alloc((size_t)BTD * 2);
  f16*   attnb = (f16*)  alloc((size_t)BTD * 2);
  float* kc    = (float*)alloc((size_t)BB * SS * DM * 4);
  float* vc    = (float*)alloc((size_t)BB * SS * DM * 4);
  float* Mm    = (float*)alloc((size_t)BB * NH * DH * DH * 4);
  f16*   h1b   = (f16*)  alloc((size_t)BB * TT * FFD * 2);
  f16*   wb    = (f16*)  alloc((size_t)FFD * DM * 2);

  const int TPB = 256;
  const int gBTD = (BTD + TPB - 1) / TPB;

  auto cvt = [&](const float* w, f16* dst, int n) {
    k_cvt_f16<<<(n + TPB - 1) / TPB, TPB, 0, stream>>>(w, dst, n);
  };
  auto gemm = [&](const f16* A, const f16* Wt, const float* bias,
                  float* Cf, f16* Ch, int M, int N, int K, int relu) {
    dim3 g(N / 128, M / 128);
    k_gemm_f16<<<g, 256, 0, stream>>>(A, Wt, bias, Cf, Ch, M, N, K, relu);
  };

  const int M = BB * TT;           // 16000 rows, divisible by 128

  k_bdt_to_btd<<<gBTD, TPB, 0, stream>>>(src, xf);
  k_bdt_to_btd<<<gBTD, TPB, 0, stream>>>(pos, posf);

  for (int l = 0; l < LAY; ++l) {
    const float* Wq_l = Wq + (size_t)l * DM * DM;
    const float* Wk_l = Wk + (size_t)l * DM * DM;
    const float* Wv_l = Wv + (size_t)l * DM * DM;
    const float* Wo_l = Wo + (size_t)l * DM * DM;
    const float* bo_l = bo + (size_t)l * DM;
    const float* cw_l = conv_w + (size_t)l * DM * DH * CF;
    const float* cb_l = conv_b + (size_t)l * DM;
    const float* nk_l = null_k + (size_t)l * BB * DM;
    const float* nv_l = null_v + (size_t)l * BB * DM;
    const float* W1_l = W1 + (size_t)l * FFD * DM;
    const float* b1_l = b1 + (size_t)l * FFD;
    const float* W2_l = W2 + (size_t)l * DM * FFD;
    const float* b2_l = b2 + (size_t)l * DM;

    // qk = x + pos (f16), x (f16)
    k_make_qk<<<gBTD, TPB, 0, stream>>>(xf, posf, qkb, xb);

    // Q, K, V projections (f16 WMMA GEMMs)
    cvt(Wq_l, wb, DM * DM); gemm(qkb, wb, nullptr, nullptr, qh, M, DM, DM, 0);
    cvt(Wk_l, wb, DM * DM); gemm(qkb, wb, nullptr, nullptr, kh, M, DM, DM, 0);
    cvt(Wv_l, wb, DM * DM); gemm(xb,  wb, nullptr, nullptr, vh, M, DM, DM, 0);

    // grouped conv compression + null-token prepend
    int nComp = BB * SS * DM;
    k_compress<<<(nComp + TPB - 1) / TPB, TPB, 0, stream>>>(kh, cw_l, cb_l, nk_l, kc);
    k_compress<<<(nComp + TPB - 1) / TPB, TPB, 0, stream>>>(vh, cw_l, cb_l, nv_l, vc);

    // linear attention: M = scale * K^T V per (b,h);  attn = Q @ M
    int nKtv = BB * NH * DH * DH;
    k_ktv<<<(nKtv + TPB - 1) / TPB, TPB, 0, stream>>>(kc, vc, Mm);
    k_apply_m<<<gBTD, TPB, 0, stream>>>(qh, Mm, attnb);

    // output projection (f32 out for residual)
    cvt(Wo_l, wb, DM * DM); gemm(attnb, wb, bo_l, resid, nullptr, M, DM, DM, 0);

    // x = LN(x + attn), also emit f16 copy for the FFN
    k_add_ln<<<(M + 7) / 8, 256, 0, stream>>>(resid, xf, xb, ln1_g + l * DM, ln1_b + l * DM, M);

    // FFN: h1 = relu(x W1^T + b1) (f16), ff = h1 W2^T + b2 (f32)
    cvt(W1_l, wb, FFD * DM); gemm(xb,  wb, b1_l, nullptr, h1b, M, FFD, DM, 1);
    cvt(W2_l, wb, DM * FFD); gemm(h1b, wb, b2_l, resid, nullptr, M, DM, FFD, 0);

    // x = LN(x + ff)
    k_add_ln<<<(M + 7) / 8, 256, 0, stream>>>(resid, xf, nullptr, ln2_g + l * DM, ln2_b + l * DM, M);
  }

  k_btd_to_bdt<<<gBTD, TPB, 0, stream>>>(xf, (float*)d_out);
}
